// MoEEP_54546084659899
// MI455X (gfx1250) — matrix operand order repaired and checked by reading, not compiled, not run on hardware
//
#include <hip/hip_runtime.h>
#include <cstdint>
#include <cstddef>

// ---------------- problem constants (from reference) ----------------
#define N_TOK 32768
#define DM    512          // d_model
#define DH    1024         // d_hidden
#define NE    8            // experts
#define CAP   16384        // capacity = N / TOP_K
#define BM    64           // rows (tokens) per MLP block

typedef __attribute__((ext_vector_type(16))) _Float16 v16h;
typedef __attribute__((ext_vector_type(8)))  float    v8f;

union FragH { v16h h; uint4 q[2]; };

// ---------------- conversion kernels ----------------
__global__ void cvt_x_kernel(const float* __restrict__ x, _Float16* __restrict__ xh, int n) {
    int i = blockIdx.x * blockDim.x + threadIdx.x;
    if (i < n) xh[i] = (_Float16)x[i];
}

// W1 [E][D][H] fp32  ->  W1t [E][H][D] f16  (B-fragment friendly: n-major, k contiguous)
__global__ void cvt_w1_kernel(const float* __restrict__ W1, _Float16* __restrict__ w1t) {
    int i = blockIdx.x * blockDim.x + threadIdx.x;
    if (i >= NE * DH * DM) return;
    int d  = i % DM;
    int r  = i / DM;
    int hc = r % DH;
    int e  = r / DH;
    w1t[i] = (_Float16)W1[((size_t)e * DM + d) * DH + hc];
}

// W2 [E][H][D] fp32  ->  W2t [E][D][H] f16
__global__ void cvt_w2_kernel(const float* __restrict__ W2, _Float16* __restrict__ w2t) {
    int i = blockIdx.x * blockDim.x + threadIdx.x;
    if (i >= NE * DM * DH) return;
    int hh = i % DH;
    int r  = i / DH;
    int dc = r % DM;
    int e  = r / DM;
    w2t[i] = (_Float16)W2[((size_t)e * DH + hh) * DM + dc];
}

// ---------------- router: one wave32 per token ----------------
__global__ void router_kernel(const float* __restrict__ x,
                              const float* __restrict__ rw,   // [D][E]
                              const float* __restrict__ rb,   // [E]
                              int*   __restrict__ topIdx,     // [N][2]
                              float* __restrict__ topW,       // [N][2]
                              float* __restrict__ probsSum,   // [E]
                              unsigned* __restrict__ counts)  // [E]
{
    __shared__ float    sP[NE];
    __shared__ unsigned sC[NE];
    if (threadIdx.x < NE) { sP[threadIdx.x] = 0.f; sC[threadIdx.x] = 0u; }
    __syncthreads();

    const int lane = threadIdx.x & 31;
    const int wv   = threadIdx.x >> 5;
    const int t    = blockIdx.x * 8 + wv;

    float acc[NE];
#pragma unroll
    for (int e = 0; e < NE; ++e) acc[e] = 0.f;

    for (int d = lane; d < DM; d += 32) {
        float xv = x[(size_t)t * DM + d];
        const float4* rp = (const float4*)(rw + (size_t)d * NE);
        float4 r0 = rp[0], r1 = rp[1];
        acc[0] += xv * r0.x; acc[1] += xv * r0.y; acc[2] += xv * r0.z; acc[3] += xv * r0.w;
        acc[4] += xv * r1.x; acc[5] += xv * r1.y; acc[6] += xv * r1.z; acc[7] += xv * r1.w;
    }
#pragma unroll
    for (int off = 16; off > 0; off >>= 1) {
#pragma unroll
        for (int e = 0; e < NE; ++e) acc[e] += __shfl_xor(acc[e], off, 32);
    }

    if (lane == 0) {
        float lg[NE], p[NE];
        float mx = -1e30f;
#pragma unroll
        for (int e = 0; e < NE; ++e) { lg[e] = acc[e] + rb[e]; mx = fmaxf(mx, lg[e]); }
        float s = 0.f;
#pragma unroll
        for (int e = 0; e < NE; ++e) { p[e] = expf(lg[e] - mx); s += p[e]; }
        float inv = 1.f / s;
#pragma unroll
        for (int e = 0; e < NE; ++e) p[e] *= inv;

        int i0 = 0;
#pragma unroll
        for (int e = 1; e < NE; ++e) if (p[e] > p[i0]) i0 = e;   // strict > : lowest idx on tie (jax)
        int i1 = (i0 == 0) ? 1 : 0;
#pragma unroll
        for (int e = 0; e < NE; ++e) if (e != i0 && p[e] > p[i1]) i1 = e;

        topIdx[t * 2 + 0] = i0;  topW[t * 2 + 0] = p[i0];
        topIdx[t * 2 + 1] = i1;  topW[t * 2 + 1] = p[i1];

#pragma unroll
        for (int e = 0; e < NE; ++e) atomicAdd(&sP[e], p[e]);
        atomicAdd(&sC[i0], 1u);
        atomicAdd(&sC[i1], 1u);
    }
    __syncthreads();
    if (threadIdx.x < NE) {
        atomicAdd(&probsSum[threadIdx.x], sP[threadIdx.x]);
        atomicAdd(&counts[threadIdx.x], sC[threadIdx.x]);
    }
}

// ---------------- dispatch: one block per expert, stable order scan ----------------
__global__ void dispatch_kernel(const int*   __restrict__ topIdx,
                                const float* __restrict__ topW,
                                unsigned* __restrict__ tokOut,  // [E][CAP]
                                float*    __restrict__ wOut)    // [E][CAP]
{
    const int e   = blockIdx.x;
    const int tid = threadIdx.x;
    __shared__ int scan[1024];
    __shared__ int sBase;
    if (tid == 0) sBase = 0;
    __syncthreads();

    for (int c0 = 0; c0 < N_TOK; c0 += 1024) {
        int t  = c0 + tid;
        int i0 = topIdx[t * 2 + 0];
        int i1 = topIdx[t * 2 + 1];
        int flag = 0; float w = 0.f;
        if (i0 == e) { flag = 1; w += topW[t * 2 + 0]; }
        if (i1 == e) { flag = 1; w += topW[t * 2 + 1]; }
        scan[tid] = flag;
        __syncthreads();
        for (int off = 1; off < 1024; off <<= 1) {
            int v   = scan[tid];
            int add = (tid >= off) ? scan[tid - off] : 0;
            __syncthreads();
            scan[tid] = v + add;
            __syncthreads();
        }
        int incl  = scan[tid];
        int total = scan[1023];
        int base  = sBase;
        int slot  = base + incl - flag;
        if (flag && slot < CAP) {
            tokOut[(size_t)e * CAP + slot] = (unsigned)t;
            wOut[(size_t)e * CAP + slot]   = w;
        }
        __syncthreads();
        if (tid == 0) sBase = base + total;
        __syncthreads();
    }
}

// ---------------- fused expert MLP: WMMA f16 -> f32 ----------------
__global__ __launch_bounds__(256)
void moe_mlp_kernel(const _Float16* __restrict__ xh,    // [N][DM] f16
                    const _Float16* __restrict__ w1t,   // [E][DH][DM] f16 (n-major)
                    const _Float16* __restrict__ w2t,   // [E][DM][DH] f16 (n-major)
                    const float* __restrict__ b1,       // [E][DH]
                    const float* __restrict__ b2,       // [E][DM]
                    const unsigned* __restrict__ tokIdx,// [E][CAP]
                    const float* __restrict__ wgt,      // [E][CAP]
                    float* __restrict__ outp)           // [N][DM]
{
    extern __shared__ char smem[];
    _Float16* Ash   = (_Float16*)smem;                              // [BM][DM]  64 KB
    _Float16* Hsh   = (_Float16*)(smem + BM * DM * 2);              // [BM][DH] 128 KB
    unsigned* tokSh = (unsigned*)(smem + BM * DM * 2 + BM * DH * 2);// [BM]
    float*    wSh   = (float*)(tokSh + BM);                         // [BM]

    const int e    = blockIdx.y;
    const int m0   = blockIdx.x * BM;
    const int tid  = threadIdx.x;
    const int lane = tid & 31;
    const int wv   = tid >> 5;          // wave 0..7
    const int lhalf = lane >> 4;        // 0/1
    const int lmod  = lane & 15;

    // stage this block's token ids / routing weights
    if (tid < BM) {
        unsigned t = tokIdx[(size_t)e * CAP + m0 + tid];
        tokSh[tid] = t;
        wSh[tid]   = (t < N_TOK) ? wgt[(size_t)e * CAP + m0 + tid] : 0.f;
    }
    __syncthreads();

    // gather A rows into LDS (4 threads per row, 16B chunks)
    {
        int row  = tid >> 2;
        int part = tid & 3;
        unsigned t = tokSh[row];
        const uint4* src = (t < N_TOK) ? (const uint4*)(xh + (size_t)t * DM) : nullptr;
        uint4* dst = (uint4*)(Ash + (size_t)row * DM);
#pragma unroll
        for (int i = 0; i < 16; ++i) {
            int c = part * 16 + i;
            uint4 v{};
            if (src) v = src[c];
            dst[c] = v;
        }
    }
    __syncthreads();

    // ---------- GEMM1: h = gelu(A[64xDM] * W1[DMxDH] + b1) ----------
    for (int nt = wv; nt < DH / 16; nt += 8) {
        const int n0 = nt * 16;
        const float bv = b1[(size_t)e * DH + n0 + lmod];
        v8f acc[4];
#pragma unroll
        for (int mt = 0; mt < 4; ++mt)
#pragma unroll
            for (int r = 0; r < 8; ++r) acc[mt][r] = bv;

        const _Float16* brow = w1t + ((size_t)e * DH + n0 + lmod) * DM;
        for (int k0 = 0; k0 < DM; k0 += 32) {
            FragH fb;
            const uint4* bp = (const uint4*)(brow + k0 + lhalf * 16);
            __builtin_prefetch(brow + k0 + 64, 0, 1);
            fb.q[0] = bp[0];
            fb.q[1] = bp[1];
#pragma unroll
            for (int mt = 0; mt < 4; ++mt) {
                FragH fa;
                const uint4* ap = (const uint4*)(Ash + (size_t)(mt * 16 + lmod) * DM + k0 + lhalf * 16);
                fa.q[0] = ap[0];
                fa.q[1] = ap[1];
                acc[mt] = __builtin_amdgcn_wmma_f32_16x16x32_f16(
                    false, fa.h, false, fb.h, (short)0, acc[mt], false, false);
            }
        }
        // exact-erf GeLU, store f16 h-tile
#pragma unroll
        for (int mt = 0; mt < 4; ++mt)
#pragma unroll
            for (int r = 0; r < 8; ++r) {
                int m = mt * 16 + r + lhalf * 8;
                float v = acc[mt][r];
                float g = 0.5f * v * (1.0f + erff(v * 0.70710678118654752f));
                Hsh[(size_t)m * DH + n0 + lmod] = (_Float16)g;
            }
    }
    __syncthreads();

    // ---------- GEMM2: out = (h[64xDH] * W2[DHxDM] + b2) * w, scatter-add ----------
    for (int nt = wv; nt < DM / 16; nt += 8) {
        const int n0 = nt * 16;
        const float bv = b2[(size_t)e * DM + n0 + lmod];
        v8f acc[4];
#pragma unroll
        for (int mt = 0; mt < 4; ++mt)
#pragma unroll
            for (int r = 0; r < 8; ++r) acc[mt][r] = bv;

        const _Float16* brow = w2t + ((size_t)e * DM + n0 + lmod) * DH;
        for (int k0 = 0; k0 < DH; k0 += 32) {
            FragH fb;
            const uint4* bp = (const uint4*)(brow + k0 + lhalf * 16);
            __builtin_prefetch(brow + k0 + 64, 0, 1);
            fb.q[0] = bp[0];
            fb.q[1] = bp[1];
#pragma unroll
            for (int mt = 0; mt < 4; ++mt) {
                FragH fa;
                const uint4* ap = (const uint4*)(Hsh + (size_t)(mt * 16 + lmod) * DH + k0 + lhalf * 16);
                fa.q[0] = ap[0];
                fa.q[1] = ap[1];
                acc[mt] = __builtin_amdgcn_wmma_f32_16x16x32_f16(
                    false, fa.h, false, fb.h, (short)0, acc[mt], false, false);
            }
        }
#pragma unroll
        for (int mt = 0; mt < 4; ++mt)
#pragma unroll
            for (int r = 0; r < 8; ++r) {
                int m = mt * 16 + r + lhalf * 8;
                unsigned t = tokSh[m];
                if (t < N_TOK) {
                    float scaled = acc[mt][r] * wSh[m];
                    atomicAdd(&outp[(size_t)t * DM + n0 + lmod], scaled);
                }
            }
    }
}

// ---------------- aux loss ----------------
__global__ void aux_kernel(const float* __restrict__ probsSum,
                           const unsigned* __restrict__ counts,
                           float* __restrict__ out_aux)
{
    float bal = 0.f, imp = 0.f;
#pragma unroll
    for (int e = 0; e < NE; ++e) {
        float density = probsSum[e] / (float)N_TOK;
        float usage   = (float)counts[e] / (float)N_TOK;
        bal += density * usage;
        imp += probsSum[e] * probsSum[e];
    }
    *out_aux = bal * (float)NE + imp / (float)NE;
}

// ---------------- host launcher ----------------
extern "C" void kernel_launch(void* const* d_in, const int* in_sizes, int n_in,
                              void* d_out, int out_size, void* d_ws, size_t ws_size,
                              hipStream_t stream) {
    (void)in_sizes; (void)n_in; (void)out_size; (void)ws_size;
    const float* x        = (const float*)d_in[0];
    const float* router_w = (const float*)d_in[1];
    const float* router_b = (const float*)d_in[2];
    const float* W1       = (const float*)d_in[3];
    const float* b1       = (const float*)d_in[4];
    const float* W2       = (const float*)d_in[5];
    const float* b2       = (const float*)d_in[6];
    float* outp = (float*)d_out;

    // workspace layout (256B-aligned offsets)
    char* ws = (char*)d_ws;
    size_t off = 0;
    auto take = [&](size_t bytes) { char* p = ws + off; off += (bytes + 255) & ~(size_t)255; return p; };
    _Float16* xh      = (_Float16*)take((size_t)N_TOK * DM * 2);   // 32 MB
    _Float16* w1t     = (_Float16*)take((size_t)NE * DH * DM * 2); //  8 MB
    _Float16* w2t     = (_Float16*)take((size_t)NE * DM * DH * 2); //  8 MB
    unsigned* tokIdx  = (unsigned*)take((size_t)NE * CAP * 4);
    float*    wgt     = (float*)   take((size_t)NE * CAP * 4);
    int*      topIdx  = (int*)     take((size_t)N_TOK * 2 * 4);
    float*    topW    = (float*)   take((size_t)N_TOK * 2 * 4);
    float*    probsSum= (float*)   take(NE * 4);
    unsigned* counts  = (unsigned*)take(NE * 4);

    // zero / init (graph-capturable memsets)
    hipMemsetAsync(d_out, 0, ((size_t)N_TOK * DM + 1) * 4, stream);
    hipMemsetAsync(probsSum, 0, NE * 4, stream);
    hipMemsetAsync(counts,   0, NE * 4, stream);
    hipMemsetAsync(tokIdx, 0xFF, (size_t)NE * CAP * 4, stream);  // 0xFFFFFFFF >= N -> invalid
    hipMemsetAsync(wgt,    0,    (size_t)NE * CAP * 4, stream);

    // conversions
    {
        int n = N_TOK * DM;
        cvt_x_kernel<<<(n + 255) / 256, 256, 0, stream>>>(x, xh, n);
        int nw = NE * DM * DH;
        cvt_w1_kernel<<<(nw + 255) / 256, 256, 0, stream>>>(W1, w1t);
        cvt_w2_kernel<<<(nw + 255) / 256, 256, 0, stream>>>(W2, w2t);
    }

    // router (8 tokens per 256-thread block)
    router_kernel<<<N_TOK / 8, 256, 0, stream>>>(x, router_w, router_b,
                                                 topIdx, topW, probsSum, counts);

    // dispatch (1 block per expert)
    dispatch_kernel<<<NE, 1024, 0, stream>>>(topIdx, topW, tokIdx, wgt);

    // fused expert MLP
    constexpr size_t SMEM = (size_t)BM * DM * 2 + (size_t)BM * DH * 2 + (size_t)BM * 8;
    hipFuncSetAttribute((const void*)moe_mlp_kernel,
                        hipFuncAttributeMaxDynamicSharedMemorySize, (int)SMEM);
    dim3 grid(CAP / BM, NE);
    moe_mlp_kernel<<<grid, 256, SMEM, stream>>>(xh, w1t, w2t, b1, b2, tokIdx, wgt, outp);

    // aux loss scalar at d_out[N*DM]
    aux_kernel<<<1, 1, 0, stream>>>(probsSum, counts, outp + (size_t)N_TOK * DM);
}
